// DeformableTaskExpertS4RForMIM_24773371363686
// MI455X (gfx1250) — matrix operand order repaired and checked by reading, not compiled, not run on hardware
//
#include <hip/hip_runtime.h>
#include <hip/hip_bf16.h>

// Multi-scale deformable attention encoder for gfx1250 (MI455X).
// GEMMs run on v_wmma_f32_16x16x32_bf16 (bf16 in, fp32 accumulate) with
// double-buffered LDS staging. Softmax / bilinear sampling / LayerNorm stay
// fp32 (wave32 shuffles).

#define LENQ   7680
#define BQ     4
#define MQ     (BQ * LENQ)   // 30720 rows
#define DM     256
#define HEADS  8
#define HD     32
#define NLVL   4
#define NPTS   4
#define FFD    512

typedef __attribute__((ext_vector_type(8)))  float  v8f;
typedef __attribute__((ext_vector_type(16))) __bf16 v16bf;
typedef __attribute__((ext_vector_type(8)))  __bf16 v8bf;

union V16U { v16bf v; v8bf h[2]; };

// ---------------------------------------------------------------------------
// Tiled WMMA GEMM:  C[M,N] = (A[M,K] (+ addvec[row%LENQ, :])) @ W[K,N] + bias
// BM=128, BN=64, BK=32. 256 threads = 8 waves arranged 4(m) x 2(n); each wave
// computes a 32x32 macro-tile via 2x2 v_wmma_f32_16x16x32_bf16.
// LDS is ping-pong double buffered: one barrier per k-step.
// ---------------------------------------------------------------------------
#define BM 128
#define BN 64
#define BK 32
#define AS 40   // LDS row stride (bf16 elems) for A  (multiple of 8 -> 16B align)
#define BS 40   // LDS row stride (bf16 elems) for B^T

__global__ __launch_bounds__(256)
void gemm_bf16_wmma(const float* __restrict__ A, const float* __restrict__ W,
                    const float* __restrict__ bias, const float* __restrict__ addvec,
                    float* __restrict__ C, int N, int K, int relu)
{
    __shared__ __bf16 aS[2][BM * AS];
    __shared__ __bf16 bS[2][BN * BS];

    const int tid  = threadIdx.x;
    const int lane = tid & 31;
    const int wv   = tid >> 5;
    const int wm   = wv & 3;          // 0..3 (m direction)
    const int wn   = wv >> 2;         // 0..1 (n direction)
    const int row0 = blockIdx.y * BM;
    const int col0 = blockIdx.x * BN;

    // ---- loop-invariant staging addresses ---------------------------------
    const int am  = tid & 127;        // A tile row
    const int akc = tid >> 7;         // 0..1 -> which 16-float chunk of BK
    const float* pAg = A + (size_t)(row0 + am) * K + akc * 16;
    // compile-time LENQ -> mul/shift, no runtime division
    const float* pPg = addvec
        ? addvec + (size_t)((row0 + am) % LENQ) * K + akc * 16 : nullptr;

    const int bn  = tid & 63;         // B^T tile row (output col)
    const int bkc = tid >> 6;         // 0..3 -> 8 k-values each
    const float* pWg = W + (size_t)(bkc * 8) * N + col0 + bn;

    const int aofs = am * AS + akc * 16;
    const int bofs = bn * BS + bkc * 8;

    v8f acc[2][2] = {};
    const int hi  = lane >> 4;        // half-wave select
    const int l15 = lane & 15;
    const int nk  = K / BK;

    float4 aReg[4];
    float  bReg[8];

    auto loadg = [&](int kb) {
        const float* pA = pAg + kb * BK;
        #pragma unroll
        for (int j = 0; j < 4; ++j) aReg[j] = ((const float4*)pA)[j];
        if (pPg) {
            const float* pP = pPg + kb * BK;
            #pragma unroll
            for (int j = 0; j < 4; ++j) {
                float4 p = ((const float4*)pP)[j];
                aReg[j].x += p.x; aReg[j].y += p.y;
                aReg[j].z += p.z; aReg[j].w += p.w;
            }
        }
        const float* pW = pWg + (size_t)kb * BK * N;
        #pragma unroll
        for (int j = 0; j < 8; ++j) bReg[j] = pW[(size_t)j * N];
    };
    auto stage = [&](int buf) {
        __bf16* dA = &aS[buf][aofs];
        #pragma unroll
        for (int j = 0; j < 4; ++j) {
            dA[4 * j + 0] = (__bf16)aReg[j].x;
            dA[4 * j + 1] = (__bf16)aReg[j].y;
            dA[4 * j + 2] = (__bf16)aReg[j].z;
            dA[4 * j + 3] = (__bf16)aReg[j].w;
        }
        __bf16* dB = &bS[buf][bofs];
        #pragma unroll
        for (int j = 0; j < 8; ++j) dB[j] = (__bf16)bReg[j];
    };

    loadg(0);
    stage(0);
    __syncthreads();

    for (int kb = 0; kb < nk; ++kb) {
        const int cur = kb & 1;
        if (kb + 1 < nk) loadg(kb + 1);   // prefetch next tile into registers

        // ---- fragments (ISA 16-bit A 16x32 / B 32x16 wave32 layout) -------
        V16U aF[2], bF[2];
        #pragma unroll
        for (int i = 0; i < 2; ++i) {
            const __bf16* pa = &aS[cur][(wm * 32 + i * 16 + l15) * AS];
            aF[i].h[0] = *(const v8bf*)(pa + 8 * hi);        // K 0..7  / 8..15
            aF[i].h[1] = *(const v8bf*)(pa + 16 + 8 * hi);   // K 16..23 / 24..31
        }
        #pragma unroll
        for (int j = 0; j < 2; ++j) {
            const __bf16* pb = &bS[cur][(wn * 32 + j * 16 + l15) * BS + 8 * hi];
            bF[j].h[0] = *(const v8bf*)(pb);
            bF[j].h[1] = *(const v8bf*)(pb + 16);
        }

        #pragma unroll
        for (int i = 0; i < 2; ++i)
            #pragma unroll
            for (int j = 0; j < 2; ++j)
                acc[i][j] = __builtin_amdgcn_wmma_f32_16x16x32_bf16(
                    false, aF[i].v, false, bF[j].v, (short)0, acc[i][j], false, false);

        if (kb + 1 < nk) stage(cur ^ 1);  // fill the other buffer
        __syncthreads();
    }

    // ---- epilogue: bias (+ReLU) + store ------------------------------------
    #pragma unroll
    for (int i = 0; i < 2; ++i) {
        #pragma unroll
        for (int j = 0; j < 2; ++j) {
            const int cN = col0 + wn * 32 + j * 16 + l15;
            const float bb = bias ? bias[cN] : 0.0f;
            #pragma unroll
            for (int v = 0; v < 8; ++v) {
                const int rM = row0 + wm * 32 + i * 16 + v + 8 * hi;
                float val = acc[i][j][v] + bb;
                if (relu) val = fmaxf(val, 0.0f);
                C[(size_t)rM * N + cN] = val;
            }
        }
    }
}

// ---------------------------------------------------------------------------
// Softmax over 16 logits per (b, token, head)
// ---------------------------------------------------------------------------
__global__ __launch_bounds__(256)
void softmax16(float* __restrict__ a, int rows)
{
    const int i = blockIdx.x * 256 + threadIdx.x;
    if (i >= rows) return;
    float* p = a + (size_t)i * 16;
    float4 v[4];
    #pragma unroll
    for (int j = 0; j < 4; ++j) v[j] = ((float4*)p)[j];
    float mx = -1e30f;
    #pragma unroll
    for (int j = 0; j < 4; ++j)
        mx = fmaxf(mx, fmaxf(fmaxf(v[j].x, v[j].y), fmaxf(v[j].z, v[j].w)));
    float s = 0.0f;
    #pragma unroll
    for (int j = 0; j < 4; ++j) {
        v[j].x = __expf(v[j].x - mx); v[j].y = __expf(v[j].y - mx);
        v[j].z = __expf(v[j].z - mx); v[j].w = __expf(v[j].w - mx);
        s += v[j].x + v[j].y + v[j].z + v[j].w;
    }
    const float r = 1.0f / s;
    #pragma unroll
    for (int j = 0; j < 4; ++j) {
        v[j].x *= r; v[j].y *= r; v[j].z *= r; v[j].w *= r;
        ((float4*)p)[j] = v[j];
    }
}

// ---------------------------------------------------------------------------
// Deformable bilinear sampling: 64 threads per (b,q); lane = (head, 4-chunk).
// value [B,LEN,H,HD], off [B,LEN,H,LVL,P,2], attn [B,LEN,H,LVL*P] -> out [B,LEN,D]
// ---------------------------------------------------------------------------
__global__ __launch_bounds__(256)
void msda_sample(const float* __restrict__ value, const float* __restrict__ off,
                 const float* __restrict__ attn, float* __restrict__ out)
{
    const int Hh[4] = {128, 96, 64, 32};
    const int Ww[4] = {32, 24, 16, 8};
    const int St[4] = {0, 4096, 6400, 7424};

    const int tid = threadIdx.x;
    const int sub = tid >> 6;            // 4 queries per block
    const int lt  = tid & 63;
    const int h   = lt >> 3;             // head 0..7
    const int ch  = lt & 7;              // 4-float chunk of HD

    const int qb = blockIdx.x * 4 + sub; // 0..30719
    const int q  = qb % LENQ;            // compile-time divisor
    const int b  = qb / LENQ;

    const int lq = (q >= 7424) ? 3 : (q >= 6400) ? 2 : (q >= 4096) ? 1 : 0;
    const int iw = Ww[lq];
    const int ii = q - St[lq];
    const float refx = ((ii % iw) + 0.5f) / (float)iw;
    const float refy = ((ii / iw) + 0.5f) / (float)Hh[lq];

    const float* offp = off  + (size_t)qb * 512 + h * 64;
    const float* atp  = attn + (size_t)qb * 128 + h * 16;

    float4 acc = make_float4(0.f, 0.f, 0.f, 0.f);

    #pragma unroll
    for (int l = 0; l < NLVL; ++l) {
        const int lh = Hh[l], lw = Ww[l];
        const float* vl = value + ((size_t)b * LENQ + St[l]) * DM + h * HD + ch * 4;
        #pragma unroll
        for (int p = 0; p < NPTS; ++p) {
            const float ox = offp[l * 16 + p * 2 + 0];
            const float oy = offp[l * 16 + p * 2 + 1];
            const float aw = atp[l * 4 + p];
            const float x = (refx + ox / (float)lw) * (float)lw - 0.5f;
            const float y = (refy + oy / (float)lh) * (float)lh - 0.5f;
            const float x0f = floorf(x), y0f = floorf(y);
            const float lx = x - x0f, ly = y - y0f;
            const int x0 = (int)x0f, y0 = (int)y0f;

            auto corner = [&](int yi, int xi, float wgt) {
                if (xi >= 0 && xi < lw && yi >= 0 && yi < lh) {
                    const float4 g = *(const float4*)(vl + (size_t)(yi * lw + xi) * DM);
                    const float ww = wgt * aw;
                    acc.x += ww * g.x; acc.y += ww * g.y;
                    acc.z += ww * g.z; acc.w += ww * g.w;
                }
            };
            corner(y0,     x0,     (1.f - lx) * (1.f - ly));
            corner(y0,     x0 + 1, lx * (1.f - ly));
            corner(y0 + 1, x0,     (1.f - lx) * ly);
            corner(y0 + 1, x0 + 1, lx * ly);
        }
    }
    *(float4*)(out + (size_t)qb * DM + h * HD + ch * 4) = acc;
}

// ---------------------------------------------------------------------------
// Fused residual + LayerNorm, one wave32 per 256-wide row (in place on x).
// ---------------------------------------------------------------------------
__global__ __launch_bounds__(256)
void add_ln(const float* __restrict__ addend, const float* __restrict__ g,
            const float* __restrict__ be, float* __restrict__ x)
{
    const int wv   = threadIdx.x >> 5;
    const int lane = threadIdx.x & 31;
    const size_t r = (size_t)blockIdx.x * 8 + wv;
    const float* xr = x + r * DM;
    const float* ar = addend + r * DM;

    float s[8];
    float sum = 0.f, sq = 0.f;
    #pragma unroll
    for (int j = 0; j < 8; ++j) {
        const int c = lane + 32 * j;
        const float v = xr[c] + ar[c];
        s[j] = v; sum += v; sq += v * v;
    }
    #pragma unroll
    for (int o = 16; o > 0; o >>= 1) {
        sum += __shfl_xor(sum, o);
        sq  += __shfl_xor(sq, o);
    }
    const float mean = sum * (1.0f / DM);
    const float var  = sq * (1.0f / DM) - mean * mean;
    const float inv  = rsqrtf(var + 1e-5f);
    #pragma unroll
    for (int j = 0; j < 8; ++j) {
        const int c = lane + 32 * j;
        x[r * DM + c] = (s[j] - mean) * inv * g[c] + be[c];
    }
}

// ---------------------------------------------------------------------------
extern "C" void kernel_launch(void* const* d_in, const int* in_sizes, int n_in,
                              void* d_out, int out_size, void* d_ws, size_t ws_size,
                              hipStream_t stream)
{
    const float* src   = (const float*)d_in[0];
    const float* pos   = (const float*)d_in[1];
    const float* Wv    = (const float*)d_in[2];
    const float* bv    = (const float*)d_in[3];
    const float* Woff  = (const float*)d_in[4];
    const float* boff  = (const float*)d_in[5];
    const float* Wattn = (const float*)d_in[6];
    const float* battn = (const float*)d_in[7];
    const float* Wout  = (const float*)d_in[8];
    const float* bout  = (const float*)d_in[9];
    const float* g1    = (const float*)d_in[10];
    const float* be1   = (const float*)d_in[11];
    const float* Wf1   = (const float*)d_in[12];
    const float* bf1   = (const float*)d_in[13];
    const float* Wf2   = (const float*)d_in[14];
    const float* bf2   = (const float*)d_in[15];
    const float* g2    = (const float*)d_in[16];
    const float* be2   = (const float*)d_in[17];

    // workspace carve-up (~195 MB total, fp32)
    float* x    = (float*)d_ws;              // MQ*256
    float* val  = x    + (size_t)MQ * DM;    // MQ*256
    float* offb = val  + (size_t)MQ * DM;    // MQ*512  (reused as FFN hidden)
    float* attn = offb + (size_t)MQ * FFD;   // MQ*128
    float* att  = attn + (size_t)MQ * 128;   // MQ*256
    float* tmp  = att  + (size_t)MQ * DM;    // MQ*256

    hipMemcpyAsync(x, src, (size_t)MQ * DM * sizeof(float),
                   hipMemcpyDeviceToDevice, stream);

    const dim3 blk(256);
    for (int i = 0; i < 4; ++i) {
        const float* Wvi  = Wv    + (size_t)i * DM * DM;
        const float* bvi  = bv    + (size_t)i * DM;
        const float* Woi  = Woff  + (size_t)i * DM * FFD;   // 256x512
        const float* boi  = boff  + (size_t)i * FFD;
        const float* Wai  = Wattn + (size_t)i * DM * 128;
        const float* bai  = battn + (size_t)i * 128;
        const float* Wui  = Wout  + (size_t)i * DM * DM;
        const float* bui  = bout  + (size_t)i * DM;
        const float* g1i  = g1    + (size_t)i * DM;
        const float* be1i = be1   + (size_t)i * DM;
        const float* Wf1i = Wf1   + (size_t)i * DM * FFD;
        const float* bf1i = bf1   + (size_t)i * FFD;
        const float* Wf2i = Wf2   + (size_t)i * FFD * DM;
        const float* bf2i = bf2   + (size_t)i * DM;
        const float* g2i  = g2    + (size_t)i * DM;
        const float* be2i = be2   + (size_t)i * DM;

        // value = x @ Wv + bv
        gemm_bf16_wmma<<<dim3(DM / BN, MQ / BM), blk, 0, stream>>>(
            x, Wvi, bvi, nullptr, val, DM, DM, 0);
        // off = (x + pos) @ Woff + boff
        gemm_bf16_wmma<<<dim3(FFD / BN, MQ / BM), blk, 0, stream>>>(
            x, Woi, boi, pos, offb, FFD, DM, 0);
        // attn logits = (x + pos) @ Wattn + battn
        gemm_bf16_wmma<<<dim3(128 / BN, MQ / BM), blk, 0, stream>>>(
            x, Wai, bai, pos, attn, 128, DM, 0);
        // softmax over LVL*P = 16
        softmax16<<<dim3((MQ * HEADS) / 256), blk, 0, stream>>>(attn, MQ * HEADS);
        // deformable bilinear sampling
        msda_sample<<<dim3(MQ / 4), blk, 0, stream>>>(val, offb, attn, att);
        // out proj
        gemm_bf16_wmma<<<dim3(DM / BN, MQ / BM), blk, 0, stream>>>(
            att, Wui, bui, nullptr, tmp, DM, DM, 0);
        // x = LN(x + attn_out)
        add_ln<<<dim3(MQ / 8), blk, 0, stream>>>(tmp, g1i, be1i, x);
        // hidden = relu(x @ Wf1 + bf1)   (reuses off buffer)
        gemm_bf16_wmma<<<dim3(FFD / BN, MQ / BM), blk, 0, stream>>>(
            x, Wf1i, bf1i, nullptr, offb, FFD, DM, 1);
        // ffn_out = hidden @ Wf2 + bf2
        gemm_bf16_wmma<<<dim3(DM / BN, MQ / BM), blk, 0, stream>>>(
            offb, Wf2i, bf2i, nullptr, tmp, DM, FFD, 0);
        // x = LN(x + ffn_out)
        add_ln<<<dim3(MQ / 8), blk, 0, stream>>>(tmp, g2i, be2i, x);
    }

    hipMemcpyAsync(d_out, x, (size_t)MQ * DM * sizeof(float),
                   hipMemcpyDeviceToDevice, stream);
}
